// CAM_43319040147556
// MI455X (gfx1250) — matrix operand order repaired and checked by reading, not compiled, not run on hardware
//
#include <hip/hip_runtime.h>

#define B_  4
#define C_  256
#define D_  32
#define H_  128
#define W_  128
#define N_  (H_*W_)   // 16384
#define XS_ 40        // LDS staging row stride (u16) — 16B-aligned rows, conflict-reduced

typedef __bf16 bf16_t;
typedef __attribute__((ext_vector_type(8)))  __bf16 v8bf;
typedef __attribute__((ext_vector_type(16))) __bf16 v16bf;
typedef __attribute__((ext_vector_type(8)))  float  v8f;
typedef unsigned short u16;
typedef unsigned int   u32;

// ---------- helpers ----------
__device__ __forceinline__ u16 f2bf(float f) {
  u32 u = __builtin_bit_cast(u32, f);
  u32 r = u + 0x7FFFu + ((u >> 16) & 1u);  // RNE
  return (u16)(r >> 16);
}

__device__ __forceinline__ v16bf mk16(v8bf lo, v8bf hi) {
  v16bf r;
#pragma unroll
  for (int i = 0; i < 8; ++i) { r[i] = lo[i]; r[i + 8] = hi[i]; }
  return r;
}

__device__ __forceinline__ v8bf zero8() {
  v8bf z;
#pragma unroll
  for (int i = 0; i < 8; ++i) z[i] = (__bf16)0.0f;
  return z;
}

// A fragment (16x32 bf16): lane = M row (lane&15), half h=lane>>4 selects K sub-blocks.
// elems 0..15 <-> K = kb + h*16 + [0..16) contiguous
__device__ __forceinline__ v16bf load_a(const u16* base, int m, long ktot, int kb, int h) {
  const u16* p = base + (long)m * ktot + kb + h * 16;
  v8bf lo = *(const v8bf*)p;
  v8bf hi = *(const v8bf*)(p + 8);
  return mk16(lo, hi);
}

// B fragment (32x16 bf16): lane = N col (lane&15); elems <-> K = kb + h*16 + [0..16) contiguous.
__device__ __forceinline__ v16bf load_b(const u16* colbase, int h) {
  const u16* p = colbase + h * 16;
  v8bf lo = *(const v8bf*)p;
  v8bf hi = *(const v8bf*)(p + 8);
  return mk16(lo, hi);
}

__device__ __forceinline__ v8f wmma_bf16(v16bf a, v16bf b, v8f c) {
  return __builtin_amdgcn_wmma_f32_16x16x32_bf16(false, a, false, b, (short)0, c, false, false);
}

__device__ __forceinline__ void store_pack8(u16* p, const v8f& v) {
  u32 w[4];
#pragma unroll
  for (int i = 0; i < 4; ++i)
    w[i] = (u32)f2bf(v[2 * i]) | ((u32)f2bf(v[2 * i + 1]) << 16);
  uint4 q; q.x = w[0]; q.y = w[1]; q.z = w[2]; q.w = w[3];
  *(uint4*)p = q;
}

__device__ __forceinline__ float rmax32(float v) {
#pragma unroll
  for (int o = 16; o; o >>= 1) v = fmaxf(v, __shfl_xor(v, o, 32));
  return v;
}
__device__ __forceinline__ float rsum32(float v) {
#pragma unroll
  for (int o = 16; o; o >>= 1) v += __shfl_xor(v, o, 32);
  return v;
}

// ---------- prep kernels ----------
// wq/wk/wv fp32 [32,256] -> bf16 [3][32][256]
__global__ void k_prep_qkvw(const float* wq, const float* wk, const float* wv, u16* out) {
  int i = blockIdx.x * blockDim.x + threadIdx.x;
  if (i >= 3 * D_ * C_) return;
  int w = i / (D_ * C_), r = i % (D_ * C_);
  const float* src = (w == 0) ? wq : (w == 1) ? wk : wv;
  out[i] = f2bf(src[r]);
}

// wp fp32 [256][32][3][3] -> bf16 [9][256][32]  (shift-major, row c with contiguous dIdx)
__global__ void k_prep_wp(const float* wp, u16* out) {
  int i = blockIdx.x * blockDim.x + threadIdx.x;
  if (i >= 9 * C_ * D_) return;
  int s = i / (C_ * D_), rem = i % (C_ * D_);
  int c = rem / D_, dd = rem % D_;
  int ky = s / 3, kx = s % 3;
  out[i] = f2bf(wp[((c * D_ + dd) * 3 + ky) * 3 + kx]);
}

// ---------- QKV projection with fused NCHW->pixel-major transpose through LDS ----------
// Q,K -> bf16 [B][32][N]; V -> bf16 transposed [B][N][32]
__global__ void __launch_bounds__(256) k_qkv(const float* __restrict__ x, const u16* __restrict__ wqkvR,
                                             const float* bq, const float* bk, const float* bv,
                                             u16* Qbf, u16* Kbf, u16* Vt) {
  __shared__ u16 wlds[3 * D_ * C_];       // 48 KB: all three weight matrices
  __shared__ u16 xlds[8][16 * XS_];       // 10 KB: per-wave 16-pixel x 32-channel staging
  int tid = threadIdx.x;
  {
    uint4* dst = (uint4*)wlds;
    const uint4* src = (const uint4*)wqkvR;
    for (int i = tid; i < (3 * D_ * C_) / 8; i += 256) dst[i] = src[i];
  }
  __syncthreads();

  int b = blockIdx.y;
  int wave = tid >> 5, lane = tid & 31;
  int h = lane >> 4, l15 = lane & 15;
  int n0 = blockIdx.x * 256 + wave * 32;   // 32 pixels (two 16-col tiles) per wave
  const float* bias[3] = {bq, bk, bv};
  u16* myx = &xlds[wave][0];

  v8f acc[3][2][2];                         // [w][mtile][ttile]
#pragma unroll
  for (int w = 0; w < 3; ++w)
#pragma unroll
    for (int m = 0; m < 2; ++m)
#pragma unroll
      for (int t = 0; t < 2; ++t)
#pragma unroll
        for (int r = 0; r < 8; ++r) acc[w][m][t][r] = 0.0f;

#pragma unroll 1
  for (int k = 0; k < 8; ++k) {
    int kb = k * 32;
    // hoist the 6 weight A-fragments for this K window (feed 12 WMMAs)
    v16bf af[3][2];
#pragma unroll
    for (int w = 0; w < 3; ++w)
#pragma unroll
      for (int m = 0; m < 2; ++m)
        af[w][m] = load_a(wlds + w * (D_ * C_), m * 16 + l15, C_, kb, h);

    // lane owns channel c = kb+lane; x row is contiguous along pixels
    const float* px = x + ((size_t)b * C_ + kb + lane) * N_ + n0;
#pragma unroll
    for (int t = 0; t < 2; ++t) {
      // stage 16 pixels x 32 channels into LDS (transpose happens here);
      // per-wave DS ops are in-order, no barrier needed
      float4 f[4];
#pragma unroll
      for (int i = 0; i < 4; ++i) f[i] = ((const float4*)(px + t * 16))[i];
#pragma unroll
      for (int nn = 0; nn < 16; ++nn)
        myx[nn * XS_ + lane] = f2bf(((const float*)f)[nn]);

      v16bf bf;
      {
        const u16* p = myx + l15 * XS_ + h * 16;   // 16 channel-contiguous bf16
        v8bf lo = *(const v8bf*)p;
        v8bf hi = *(const v8bf*)(p + 8);
        bf = mk16(lo, hi);
      }
#pragma unroll
      for (int w = 0; w < 3; ++w)
#pragma unroll
        for (int m = 0; m < 2; ++m)
          acc[w][m][t] = wmma_bf16(af[w][m], bf, acc[w][m][t]);
    }
  }

  // epilogue: bias + stores
#pragma unroll
  for (int t = 0; t < 2; ++t) {
    int n = n0 + t * 16 + l15;
#pragma unroll
    for (int m = 0; m < 2; ++m) {
      // Q, K: row-major bf16 [32][N]
#pragma unroll
      for (int r = 0; r < 8; ++r) {
        int dd = m * 16 + 8 * h + r;
        Qbf[((size_t)b * D_ + dd) * N_ + n] = f2bf(acc[0][m][t][r] + bias[0][dd]);
        Kbf[((size_t)b * D_ + dd) * N_ + n] = f2bf(acc[1][m][t][r] + bias[1][dd]);
      }
      // V: transposed [N][32], 8 contiguous bf16 per lane -> one b128 store
      v8f vv;
#pragma unroll
      for (int r = 0; r < 8; ++r) vv[r] = acc[2][m][t][r] + bias[2][m * 16 + 8 * h + r];
      store_pack8(Vt + ((size_t)b * N_ + n) * D_ + m * 16 + h * 8, vv);
    }
  }
}

// ---------- energy = Q*K^T [B][32][32], split over N with fp32 atomics ----------
__global__ void __launch_bounds__(256) k_energy(const u16* __restrict__ Qbf, const u16* __restrict__ Kbf,
                                                float* energy) {
  int b = blockIdx.y;
  int tid = threadIdx.x;
  int wave = tid >> 5, lane = tid & 31;
  int h = lane >> 4, l15 = lane & 15;
  int widx = blockIdx.x * 8 + wave;        // 0..63
  int nbase = widx * (N_ / 64);            // 256 columns per wave

  v8f acc[2][2];
#pragma unroll
  for (int m = 0; m < 2; ++m)
#pragma unroll
    for (int e = 0; e < 2; ++e)
#pragma unroll
      for (int r = 0; r < 8; ++r) acc[m][e][r] = 0.0f;

  const u16* qb = Qbf + (size_t)b * D_ * N_;
  const u16* kbm = Kbf + (size_t)b * D_ * N_;
#pragma unroll 1
  for (int k = 0; k < 8; ++k) {
    int kb = nbase + k * 32;
    v16bf af[2], bf[2];
#pragma unroll
    for (int m = 0; m < 2; ++m) af[m] = load_a(qb, m * 16 + l15, N_, kb, h);
#pragma unroll
    for (int e = 0; e < 2; ++e) bf[e] = load_b(kbm + (size_t)(e * 16 + l15) * N_ + kb, h);
#pragma unroll
    for (int m = 0; m < 2; ++m)
#pragma unroll
      for (int e = 0; e < 2; ++e) acc[m][e] = wmma_bf16(af[m], bf[e], acc[m][e]);
  }
#pragma unroll
  for (int m = 0; m < 2; ++m)
#pragma unroll
    for (int e = 0; e < 2; ++e)
#pragma unroll
      for (int r = 0; r < 8; ++r) {
        int row = m * 16 + 8 * h + r;
        int col = e * 16 + l15;
        atomicAdd(&energy[b * 1024 + row * 32 + col], acc[m][e][r]);
      }
}

// ---------- softmax(max(row) - energy) -> attention bf16 [B][32][32] ----------
__global__ void k_softmax(const float* __restrict__ energy, u16* attbf) {
  int b = blockIdx.x;
  int e = threadIdx.x, dd = threadIdx.y;   // blockDim (32,32): each y-row is one wave
  float val = energy[b * 1024 + dd * 32 + e];
  float mx = rmax32(val);
  float nv = mx - val;
  float mx2 = rmax32(nv);
  float ex = expf(nv - mx2);
  float s = rsum32(ex);
  attbf[b * 1024 + dd * 32 + e] = f2bf(ex / s);
}

// ---------- O1 = att * V, written transposed bf16 [B][N][32] ----------
__global__ void __launch_bounds__(256) k_attv(const u16* __restrict__ attbf, const u16* __restrict__ Vt,
                                              u16* O1t) {
  int b = blockIdx.y;
  int tid = threadIdx.x;
  int wave = tid >> 5, lane = tid & 31;
  int h = lane >> 4, l15 = lane & 15;
  int widx = blockIdx.x * 8 + wave;        // 0..255, 4 tiles each

  v16bf af[2];
#pragma unroll
  for (int m = 0; m < 2; ++m) af[m] = load_a(attbf + b * 1024, m * 16 + l15, 32, 0, h);

#pragma unroll 1
  for (int t = 0; t < 4; ++t) {
    int n = (widx * 4 + t) * 16 + l15;
    v16bf bf = load_b(Vt + ((size_t)b * N_ + n) * D_, h);
#pragma unroll
    for (int m = 0; m < 2; ++m) {
      v8f acc;
#pragma unroll
      for (int r = 0; r < 8; ++r) acc[r] = 0.0f;
      acc = wmma_bf16(af[m], bf, acc);
      store_pack8(O1t + ((size_t)b * N_ + n) * D_ + m * 16 + h * 8, acc);
    }
  }
}

// ---------- 3x3 conv (9 shifted K=32 WMMA steps) + exact GELU + residual ----------
__global__ void __launch_bounds__(256) k_conv(const u16* __restrict__ O1t, const u16* __restrict__ wpR,
                                              const float* __restrict__ x, const float* __restrict__ gamma,
                                              float* __restrict__ out) {
  int b = blockIdx.z;
  int c0 = blockIdx.y * 16;
  int tid = threadIdx.x;
  int wave = tid >> 5, lane = tid & 31;
  int h = lane >> 4, l15 = lane & 15;

  v16bf af[9];
#pragma unroll
  for (int s = 0; s < 9; ++s) af[s] = load_a(wpR + s * (C_ * D_), c0 + l15, D_, 0, h);

  const float gm = gamma[0];
  const u16* o1b = O1t + (size_t)b * N_ * D_;

#pragma unroll 1
  for (int t = 0; t < 8; ++t) {
    int nt = blockIdx.x * 64 + wave * 8 + t;   // 0..1023
    int n0 = nt * 16;
    int hr = n0 >> 7;            // image row (W=128, 16-pixel tiles stay in-row)
    int wc = (n0 & 127) + l15;   // per-lane image column

    // prefetch next tile's activation + residual lines (global_prefetch_b8)
    if (t < 7) {
      __builtin_prefetch(o1b + (size_t)(n0 + 16) * D_ + lane * 16, 0, 1);
      __builtin_prefetch(x + ((size_t)b * C_ + c0 + 8 * h) * N_ + n0 + 16 + l15, 0, 1);
    }

    v8f acc;
#pragma unroll
    for (int r = 0; r < 8; ++r) acc[r] = 0.0f;

#pragma unroll
    for (int s = 0; s < 9; ++s) {
      int dy = s / 3 - 1, dx = s % 3 - 1;
      int hs = hr + dy;
      if (hs < 0 || hs >= H_) continue;       // wave-uniform skip
      int ws = wc + dx;
      bool valid = (ws >= 0) && (ws < W_);
      int pix = hs * W_ + (valid ? ws : 0);   // clamp for safe load
      const u16* p = o1b + (size_t)pix * D_ + h * 16;
      v8bf lo = *(const v8bf*)p;
      v8bf hi = *(const v8bf*)(p + 8);
      if (!valid) { lo = zero8(); hi = zero8(); }
      acc = wmma_bf16(af[s], mk16(lo, hi), acc);
    }
    // epilogue: GELU(exact) * gamma + x
#pragma unroll
    for (int r = 0; r < 8; ++r) {
      int c = c0 + 8 * h + r;
      int n = n0 + l15;
      size_t idx = ((size_t)b * C_ + c) * N_ + n;
      float v = acc[r];
      float g = 0.5f * v * (1.0f + erff(v * 0.70710678118654752f));
      out[idx] = gm * g + x[idx];
    }
  }
}

// ---------- host launch ----------
extern "C" void kernel_launch(void* const* d_in, const int* in_sizes, int n_in,
                              void* d_out, int out_size, void* d_ws, size_t ws_size,
                              hipStream_t stream) {
  const float* x  = (const float*)d_in[0];
  const float* wq = (const float*)d_in[1];
  const float* bq = (const float*)d_in[2];
  const float* wk = (const float*)d_in[3];
  const float* bk = (const float*)d_in[4];
  const float* wv = (const float*)d_in[5];
  const float* bv = (const float*)d_in[6];
  const float* wp = (const float*)d_in[7];
  const float* gamma = (const float*)d_in[8];
  float* out = (float*)d_out;

  char* w = (char*)d_ws;
  u16* Qbf   = (u16*)w;  w += (size_t)B_ * D_ * N_ * 2;   // 4 MB
  u16* Kbf   = (u16*)w;  w += (size_t)B_ * D_ * N_ * 2;   // 4 MB
  u16* Vt    = (u16*)w;  w += (size_t)B_ * N_ * D_ * 2;   // 4 MB
  u16* O1t   = (u16*)w;  w += (size_t)B_ * N_ * D_ * 2;   // 4 MB
  u16* wqkvR = (u16*)w;  w += (size_t)3 * D_ * C_ * 2;    // 48 KB
  u16* wpR   = (u16*)w;  w += (size_t)9 * C_ * D_ * 2;    // 144 KB
  float* energy = (float*)w; w += (size_t)B_ * 1024 * 4;  // 16 KB
  u16* attbf = (u16*)w;  w += (size_t)B_ * 1024 * 2;      // 8 KB

  k_prep_qkvw<<<(3 * D_ * C_ + 255) / 256, 256, 0, stream>>>(wq, wk, wv, wqkvR);
  k_prep_wp<<<(9 * C_ * D_ + 255) / 256, 256, 0, stream>>>(wp, wpR);
  hipMemsetAsync(energy, 0, (size_t)B_ * 1024 * 4, stream);
  k_qkv<<<dim3(64, B_), 256, 0, stream>>>(x, wqkvR, bq, bk, bv, Qbf, Kbf, Vt);
  k_energy<<<dim3(8, B_), 256, 0, stream>>>(Qbf, Kbf, energy);
  k_softmax<<<B_, dim3(32, 32), 0, stream>>>(energy, attbf);
  k_attv<<<dim3(32, B_), 256, 0, stream>>>(attbf, Vt, O1t);
  k_conv<<<dim3(16, 16, B_), 256, 0, stream>>>(O1t, wpR, x, gamma, out);
}
